// Model_21242908246643
// MI455X (gfx1250) — matrix operand order repaired and checked by reading, not compiled, not run on hardware
//
#include <hip/hip_runtime.h>
#include <hip/hip_bf16.h>

typedef __attribute__((ext_vector_type(16))) __bf16 v16bf;
typedef __attribute__((ext_vector_type(8)))  __bf16 v8bf;
typedef __attribute__((ext_vector_type(8)))  float  v8f;

#define WGSIZE 256

// ---------------------------------------------------------------------------
// WMMA helpers (CDNA5: D = A(16x32 bf16) x B(32x16 bf16) + C(16x16 f32))
// ---------------------------------------------------------------------------
__device__ __forceinline__ v8f wmma_bf16(v16bf a, v16bf b, v8f c) {
  return __builtin_amdgcn_wmma_f32_16x16x32_bf16(
      /*neg_a=*/false, a, /*neg_b=*/false, b,
      /*c_mod=*/(short)0, c, /*reuse_a=*/false, /*reuse_b=*/false);
}

// A-matrix fragment from a row-major bf16 buffer (LDS): 16xK tile at row0,
// k window [k0, k0+32). Lanes 0-15: M=lane, K = k0+0..7 and k0+16..23;
// lanes 16-31: M=lane-16, K = k0+8..15 and k0+24..31.
__device__ __forceinline__ v16bf load_a_frag(const __bf16* Abuf, int row0,
                                             int ldk, int k0) {
  int lane = threadIdx.x & 31;
  int m  = lane & 15;
  int ph = lane >> 4;
  const __bf16* base = Abuf + (size_t)(row0 + m) * ldk;
  v8bf lo = *(const v8bf*)(base + k0 + ph * 8);
  v8bf hi = *(const v8bf*)(base + k0 + 16 + ph * 8);
  v16bf a;
#pragma unroll
  for (int i = 0; i < 8; ++i) { a[i] = lo[i]; a[8 + i] = hi[i]; }
  return a;
}

// B-matrix fragment from a row-major bf16 weight matrix W (Nout x K):
// B(k,n) = W[n,k]. Lane l: n = n0+(l&15), h=(l>>4): K = k0+8h+0..7, k0+16+8h+0..7.
__device__ __forceinline__ v16bf load_b_frag(const __bf16* W, int n0,
                                             int ldk, int k0) {
  int lane = threadIdx.x & 31;
  int n  = n0 + (lane & 15);
  int kb = k0 + (lane >> 4) * 8;
  const __bf16* base = W + (size_t)n * ldk + kb;
  v8bf lo = *(const v8bf*)(base);
  v8bf hi = *(const v8bf*)(base + 16);
  v16bf b;
#pragma unroll
  for (int i = 0; i < 8; ++i) { b[i] = lo[i]; b[8 + i] = hi[i]; }
  return b;
}

// ---------------------------------------------------------------------------
// Stage 64 rows of f32 activations into LDS as bf16 with optional layernorm.
// Kdim must be a multiple of 32 (64 or 512 here). 8 waves, 8 rows each.
// ---------------------------------------------------------------------------
__device__ void stage_rows(const float* A, __bf16* Abuf, int row0_global,
                           int Kdim, const float* lnw, const float* lnb,
                           int use_ln, int per_ens) {
  int wave = threadIdx.x >> 5, lane = threadIdx.x & 31;
  int nper = Kdim >> 5;
  for (int r = wave; r < 64; r += 8) {
    int grow = row0_global + r;
    const float* src = A + (size_t)grow * Kdim;
    float vals[16];
    float s = 0.f, ss = 0.f;
#pragma unroll 4
    for (int j = 0; j < nper; ++j) {
      float v = src[lane + (j << 5)];
      vals[j] = v; s += v; ss += v * v;
    }
    if (use_ln) {
      for (int o = 16; o; o >>= 1) {
        s  += __shfl_xor(s, o, 32);
        ss += __shfl_xor(ss, o, 32);
      }
      float mu   = s / (float)Kdim;
      float var  = ss / (float)Kdim - mu * mu;
      float rstd = rsqrtf(var + 1e-5f);
      const float* w = lnw;
      const float* b = lnb;
      if (per_ens) { int e = grow & 7; w += e * Kdim; b += e * Kdim; }
      for (int j = 0; j < nper; ++j) {
        int d = lane + (j << 5);
        Abuf[(size_t)r * Kdim + d] =
            (__bf16)((vals[j] - mu) * rstd * w[d] + b[d]);
      }
    } else {
      for (int j = 0; j < nper; ++j) {
        int d = lane + (j << 5);
        Abuf[(size_t)r * Kdim + d] = (__bf16)vals[j];
      }
    }
  }
}

// ---------------------------------------------------------------------------
// Generic: out(Mx512) = LN?(A(MxK)) @ W^T + bias.  64 rows per WG.
// Optionally also emit bf16(out) and per-row sum(out^2) (for k-projection).
// ---------------------------------------------------------------------------
__global__ __launch_bounds__(WGSIZE, 1)
void gemm_ln_kernel(const float* __restrict__ A,
                    const __bf16* __restrict__ W,
                    const float* __restrict__ bias,
                    const float* __restrict__ lnw,
                    const float* __restrict__ lnb,
                    float* __restrict__ out,
                    __bf16* __restrict__ out_bf,
                    float* __restrict__ ksq,
                    int Kdim, int use_ln) {
  extern __shared__ char smem[];
  __bf16* Abuf = (__bf16*)smem;
  float* rowsq = (float*)(smem + (size_t)64 * Kdim * sizeof(__bf16));
  int row0 = blockIdx.x * 64;
  stage_rows(A, Abuf, row0, Kdim, lnw, lnb, use_ln, 0);
  if (ksq && threadIdx.x < 64) rowsq[threadIdx.x] = 0.f;
  __syncthreads();

  int wave = threadIdx.x >> 5, lane = threadIdx.x & 31;
  int ncol0 = wave * 64;
  const v8f vz = {};
  v8f acc[4][4];
#pragma unroll
  for (int mt = 0; mt < 4; ++mt)
#pragma unroll
    for (int nt = 0; nt < 4; ++nt) acc[mt][nt] = vz;

  int ksteps = Kdim >> 5;
  for (int ks = 0; ks < ksteps; ++ks) {
    v16bf af[4];
#pragma unroll
    for (int mt = 0; mt < 4; ++mt)
      af[mt] = load_a_frag(Abuf, mt * 16, Kdim, ks * 32);
#pragma unroll
    for (int nt = 0; nt < 4; ++nt) {
      v16bf bf = load_b_frag(W, ncol0 + nt * 16, Kdim, ks * 32);
#pragma unroll
      for (int mt = 0; mt < 4; ++mt)
        acc[mt][nt] = wmma_bf16(af[mt], bf, acc[mt][nt]);
    }
  }

#pragma unroll
  for (int mt = 0; mt < 4; ++mt)
#pragma unroll
    for (int nt = 0; nt < 4; ++nt) {
#pragma unroll
      for (int r = 0; r < 8; ++r) {
        int m = mt * 16 + (lane >> 4) * 8 + r;
        int n = ncol0 + nt * 16 + (lane & 15);
        float v = acc[mt][nt][r] + bias[n];
        size_t o = (size_t)(row0 + m) * 512 + n;
        out[o] = v;
        if (out_bf) out_bf[o] = (__bf16)v;
        if (ksq) atomicAdd(&rowsq[m], v * v);
      }
    }
  if (ksq) {
    __syncthreads();
    if (threadIdx.x < 64) ksq[row0 + threadIdx.x] = rowsq[threadIdx.x];
  }
}

// ---------------------------------------------------------------------------
// Fused residual MLP block: H += relu(LN?(H) @ W1^T + b1) @ W2^T + b2.
// d=512, hidden=1024 processed in 128-wide chunks via LDS round-trip.
// ---------------------------------------------------------------------------
__global__ __launch_bounds__(WGSIZE, 1)
void mlp_block_kernel(float* __restrict__ H,
                      const __bf16* __restrict__ W1,
                      const float* __restrict__ b1,
                      const __bf16* __restrict__ W2,
                      const float* __restrict__ b2,
                      const float* __restrict__ lnw,
                      const float* __restrict__ lnb,
                      int use_ln, int per_ens) {
  extern __shared__ char smem[];
  __bf16* Abuf = (__bf16*)smem;                     // 64*512 bf16
  __bf16* Hbuf = (__bf16*)(smem + 64 * 512 * 2);    // 64*128 bf16
  int row0 = blockIdx.x * 64;
  stage_rows(H, Abuf, row0, 512, lnw, lnb, use_ln, per_ens);
  __syncthreads();

  int wave = threadIdx.x >> 5, lane = threadIdx.x & 31;
  const v8f vz = {};
  v8f acc[4][4];
#pragma unroll
  for (int mt = 0; mt < 4; ++mt)
#pragma unroll
    for (int nt = 0; nt < 4; ++nt) acc[mt][nt] = vz;

  int mt1 = wave & 3;   // GEMM1 m-tile for this wave
  int hg  = wave >> 2;  // GEMM1 h-tile group (0/1)

  for (int hc = 0; hc < 8; ++hc) {
    // GEMM1: t-chunk = relu(A @ W1^T + b1) for hidden cols [hc*128, +128)
    v8f acc1[4] = {vz, vz, vz, vz};
    for (int ks = 0; ks < 16; ++ks) {
      v16bf af = load_a_frag(Abuf, mt1 * 16, 512, ks * 32);
#pragma unroll
      for (int ht = 0; ht < 4; ++ht) {
        v16bf bf = load_b_frag(W1, hc * 128 + (hg * 4 + ht) * 16, 512, ks * 32);
        acc1[ht] = wmma_bf16(af, bf, acc1[ht]);
      }
    }
#pragma unroll
    for (int ht = 0; ht < 4; ++ht)
#pragma unroll
      for (int r = 0; r < 8; ++r) {
        int m  = mt1 * 16 + (lane >> 4) * 8 + r;
        int cl = (hg * 4 + ht) * 16 + (lane & 15);
        float v = acc1[ht][r] + b1[hc * 128 + cl];
        Hbuf[m * 128 + cl] = (__bf16)fmaxf(v, 0.f);
      }
    __syncthreads();
    // GEMM2: acc += t-chunk @ W2[:, hc*128:+128]^T
    for (int k2 = 0; k2 < 4; ++k2) {
      v16bf a2[4];
#pragma unroll
      for (int mt = 0; mt < 4; ++mt)
        a2[mt] = load_a_frag(Hbuf, mt * 16, 128, k2 * 32);
#pragma unroll
      for (int nt = 0; nt < 4; ++nt) {
        v16bf b2f = load_b_frag(W2, wave * 64 + nt * 16, 1024,
                                hc * 128 + k2 * 32);
#pragma unroll
        for (int mt = 0; mt < 4; ++mt)
          acc[mt][nt] = wmma_bf16(a2[mt], b2f, acc[mt][nt]);
      }
    }
    __syncthreads();
  }
#pragma unroll
  for (int mt = 0; mt < 4; ++mt)
#pragma unroll
    for (int nt = 0; nt < 4; ++nt)
#pragma unroll
      for (int r = 0; r < 8; ++r) {
        int m = mt * 16 + (lane >> 4) * 8 + r;
        int n = wave * 64 + nt * 16 + (lane & 15);
        size_t o = (size_t)(row0 + m) * 512 + n;
        H[o] = H[o] + acc[mt][nt][r] + b2[n];
      }
}

// ---------------------------------------------------------------------------
// Top-96 nearest candidates per query. 16 queries/WG, WMMA scores via
// s = ||ck||^2 - 2*kq.ck (rank-equivalent to d2). 1024-candidate chunks.
// ---------------------------------------------------------------------------
__global__ __launch_bounds__(WGSIZE, 1)
void topk_kernel(const float* __restrict__ kx,
                 const __bf16* __restrict__ ckbf,
                 const float* __restrict__ cksq,
                 int* __restrict__ ctx_idx) {
  extern __shared__ char smem[];
  __bf16* Aq   = (__bf16*)smem;                         // 16*512 bf16
  float*  S    = (float*)(smem + 16 * 512 * 2);         // 16*1024 f32
  float*  bestd = S + 16 * 1024;                        // 16*96
  int*    besti = (int*)(bestd + 16 * 96);              // 16*96
  int q0 = blockIdx.x * 16;
  for (int i = threadIdx.x; i < 16 * 512; i += WGSIZE)
    Aq[i] = (__bf16)kx[(size_t)(q0 + (i >> 9)) * 512 + (i & 511)];
  for (int i = threadIdx.x; i < 16 * 96; i += WGSIZE) bestd[i] = 3.4e38f;
  __syncthreads();

  int wave = threadIdx.x >> 5, lane = threadIdx.x & 31;
  const v8f vz = {};
  for (int chunk = 0; chunk < 64; ++chunk) {
    int c0 = chunk * 1024;
    v8f acc[8] = {vz, vz, vz, vz, vz, vz, vz, vz};
    for (int ks = 0; ks < 16; ++ks) {
      v16bf af = load_a_frag(Aq, 0, 512, ks * 32);
#pragma unroll
      for (int nt = 0; nt < 8; ++nt) {
        v16bf bf = load_b_frag(ckbf, c0 + (wave * 8 + nt) * 16, 512, ks * 32);
        acc[nt] = wmma_bf16(af, bf, acc[nt]);
      }
    }
#pragma unroll
    for (int nt = 0; nt < 8; ++nt)
#pragma unroll
      for (int r = 0; r < 8; ++r) {
        int m = (lane >> 4) * 8 + r;
        int n = (wave * 8 + nt) * 16 + (lane & 15);
        S[m * 1024 + n] = cksq[c0 + n] - 2.f * acc[nt][r];
      }
    __syncthreads();
    if (threadIdx.x < 16) {
      int q = threadIdx.x;
      float* bd = bestd + q * 96;
      int*   bi = besti + q * 96;
      for (int j = 0; j < 1024; ++j) {
        float s = S[q * 1024 + j];
        if (s < bd[95]) {
          int pos = 95;
          while (pos > 0 && bd[pos - 1] > s) {
            bd[pos] = bd[pos - 1]; bi[pos] = bi[pos - 1]; --pos;
          }
          bd[pos] = s; bi[pos] = c0 + j;
        }
      }
    }
    __syncthreads();
  }
  for (int i = threadIdx.x; i < 16 * 96; i += WGSIZE)
    ctx_idx[(size_t)q0 * 96 + i] = besti[i];
}

// ---------------------------------------------------------------------------
// Per-query context aggregation: softmax over -||k-ck||^2, T-MLP on diff,
// label embedding, prob-weighted sum, residual into h. One WG per query.
// ---------------------------------------------------------------------------
__global__ __launch_bounds__(WGSIZE, 1)
void context_kernel(const float* __restrict__ kx,
                    const float* __restrict__ ck,
                    const float* __restrict__ cy,
                    const int* __restrict__ ctx_idx,
                    const __bf16* __restrict__ TW1,
                    const float* __restrict__ Tb1,
                    const __bf16* __restrict__ TW2,
                    const float* __restrict__ labW,
                    const float* __restrict__ labb,
                    float* __restrict__ h) {  // in: x, out: x+ctx
  extern __shared__ char smem[];
  float* kq     = (float*)smem;            // 512
  float* dist   = kq + 512;                // 96
  float* probs  = dist + 96;               // 96
  float* yv     = probs + 96;              // 96
  float* ctxacc = yv + 96;                 // 512
  __bf16* diffb = (__bf16*)(ctxacc + 512); // 96*512 bf16
  __bf16* hbuf  = diffb + 96 * 512;        // 48*128 bf16
  int b = blockIdx.x, tid = threadIdx.x;

  for (int d = tid; d < 512; d += WGSIZE) {
    kq[d] = kx[(size_t)b * 512 + d];
    ctxacc[d] = 0.f;
  }
  if (tid < 96) dist[tid] = 0.f;
  __syncthreads();

  for (int c = 0; c < 96; ++c) {
    int idx = ctx_idx[b * 96 + c];
    float part = 0.f;
    for (int d = tid; d < 512; d += WGSIZE) {
      float df = kq[d] - ck[(size_t)idx * 512 + d];
      diffb[c * 512 + d] = (__bf16)df;
      part += df * df;
    }
    atomicAdd(&dist[c], part);
  }
  if (tid < 96) yv[tid] = cy[ctx_idx[b * 96 + tid]];
  __syncthreads();

  if (tid < 32) {  // softmax over sims = -dist
    float mx = -3.4e38f;
    for (int j = 0; j < 3; ++j) mx = fmaxf(mx, -dist[tid + 32 * j]);
    for (int o = 16; o; o >>= 1) mx = fmaxf(mx, __shfl_xor(mx, o, 32));
    float e[3], s = 0.f;
    for (int j = 0; j < 3; ++j) {
      e[j] = __expf(-dist[tid + 32 * j] - mx);
      s += e[j];
    }
    for (int o = 16; o; o >>= 1) s += __shfl_xor(s, o, 32);
    for (int j = 0; j < 3; ++j) probs[tid + 32 * j] = e[j] / s;
  }
  __syncthreads();

  int wave = tid >> 5, lane = tid & 31;
  const v8f vz = {};
  for (int half = 0; half < 2; ++half) {
    int r0 = half * 48;
    v8f tacc[3][4];
#pragma unroll
    for (int mt = 0; mt < 3; ++mt)
#pragma unroll
      for (int nt = 0; nt < 4; ++nt) tacc[mt][nt] = vz;

    for (int hc = 0; hc < 8; ++hc) {
      v8f acc1[3] = {vz, vz, vz};
      for (int ks = 0; ks < 16; ++ks) {
        v16bf bf = load_b_frag(TW1, hc * 128 + wave * 16, 512, ks * 32);
#pragma unroll
        for (int mt = 0; mt < 3; ++mt) {
          v16bf af = load_a_frag(diffb, r0 + mt * 16, 512, ks * 32);
          acc1[mt] = wmma_bf16(af, bf, acc1[mt]);
        }
      }
#pragma unroll
      for (int mt = 0; mt < 3; ++mt)
#pragma unroll
        for (int r = 0; r < 8; ++r) {
          int m  = mt * 16 + (lane >> 4) * 8 + r;
          int cl = wave * 16 + (lane & 15);
          float v = acc1[mt][r] + Tb1[hc * 128 + cl];
          hbuf[m * 128 + cl] = (__bf16)fmaxf(v, 0.f);
        }
      __syncthreads();
      for (int k2 = 0; k2 < 4; ++k2) {
        v16bf a2[3];
#pragma unroll
        for (int mt = 0; mt < 3; ++mt)
          a2[mt] = load_a_frag(hbuf, mt * 16, 128, k2 * 32);
#pragma unroll
        for (int nt = 0; nt < 4; ++nt) {
          v16bf b2f = load_b_frag(TW2, wave * 64 + nt * 16, 1024,
                                  hc * 128 + k2 * 32);
#pragma unroll
          for (int mt = 0; mt < 3; ++mt)
            tacc[mt][nt] = wmma_bf16(a2[mt], b2f, tacc[mt][nt]);
        }
      }
      __syncthreads();
    }
    // values = ctx_y_emb + t ; ctxacc += probs[c] * values
#pragma unroll
    for (int mt = 0; mt < 3; ++mt)
#pragma unroll
      for (int nt = 0; nt < 4; ++nt)
#pragma unroll
        for (int r = 0; r < 8; ++r) {
          int c = r0 + mt * 16 + (lane >> 4) * 8 + r;
          int n = wave * 64 + nt * 16 + (lane & 15);
          float v = tacc[mt][nt][r] + yv[c] * labW[n] + labb[n];
          atomicAdd(&ctxacc[n], probs[c] * v);
        }
    __syncthreads();
  }
  for (int d = tid; d < 512; d += WGSIZE)
    h[(size_t)b * 512 + d] = h[(size_t)b * 512 + d] + ctxacc[d];
}

// ---------------------------------------------------------------------------
__global__ void broadcast8_kernel(const float* __restrict__ h,
                                  float* __restrict__ Hp, int total) {
  int i = blockIdx.x * WGSIZE + threadIdx.x;
  if (i < total) {
    int row = i >> 9, d = i & 511, b = row >> 3;
    Hp[i] = h[(size_t)b * 512 + d];
  }
}

__global__ void cvt_bf16_kernel(const float* __restrict__ in,
                                __bf16* __restrict__ out, int n) {
  int i = blockIdx.x * WGSIZE + threadIdx.x;
  if (i < n) out[i] = (__bf16)in[i];
}

// out[row] = relu(LN(h,head_ln)) . head_W + head_b  (one wave per row)
__global__ void head_kernel(const float* __restrict__ Hp,
                            const float* __restrict__ lnw,
                            const float* __restrict__ lnb,
                            const float* __restrict__ hW,
                            const float* __restrict__ hb,
                            float* __restrict__ out) {
  int wave = threadIdx.x >> 5, lane = threadIdx.x & 31;
  int row = blockIdx.x * 8 + wave;
  const float* src = Hp + (size_t)row * 512;
  float vals[16], s = 0.f, ss = 0.f;
#pragma unroll
  for (int j = 0; j < 16; ++j) {
    float v = src[lane + 32 * j];
    vals[j] = v; s += v; ss += v * v;
  }
  for (int o = 16; o; o >>= 1) {
    s += __shfl_xor(s, o, 32); ss += __shfl_xor(ss, o, 32);
  }
  float mu = s / 512.f;
  float rstd = rsqrtf(ss / 512.f - mu * mu + 1e-5f);
  float acc = 0.f;
#pragma unroll
  for (int j = 0; j < 16; ++j) {
    int d = lane + 32 * j;
    float v = (vals[j] - mu) * rstd * lnw[d] + lnb[d];
    acc += fmaxf(v, 0.f) * hW[d];
  }
  for (int o = 16; o; o >>= 1) acc += __shfl_xor(acc, o, 32);
  if (lane == 0) out[row] = acc + hb[0];
}

// ---------------------------------------------------------------------------
// Host orchestration
// ---------------------------------------------------------------------------
extern "C" void kernel_launch(void* const* d_in, const int* in_sizes, int n_in,
                              void* d_out, int out_size, void* d_ws,
                              size_t ws_size, hipStream_t stream) {
  (void)in_sizes; (void)n_in; (void)out_size; (void)ws_size;
  const int MC = 65536, MX = 2048, MP = 16384, CTX = 96;

  const float* x_num  = (const float*)d_in[0];
  const float* cand_x = (const float*)d_in[1];
  const float* cand_y = (const float*)d_in[2];
  const float* linW   = (const float*)d_in[3];
  const float* linb   = (const float*)d_in[4];
  // enc blocks: [5..10], [11..16] = ln_w, ln_b, W1, b1, W2, b2
  const float* mixlnw = (const float*)d_in[17];
  const float* mixlnb = (const float*)d_in[18];
  const float* KW     = (const float*)d_in[19];
  const float* Kb     = (const float*)d_in[20];
  const float* TW1    = (const float*)d_in[21];
  const float* Tb1    = (const float*)d_in[22];
  const float* TW2    = (const float*)d_in[23];
  const float* labW   = (const float*)d_in[24];
  const float* labb   = (const float*)d_in[25];
  // pred blocks: [26..31], [32..37] = pln_w, pln_b, W1, b1, W2, b2
  const float* hlnw = (const float*)d_in[38];
  const float* hlnb = (const float*)d_in[39];
  const float* hW   = (const float*)d_in[40];
  const float* hb   = (const float*)d_in[41];

  char* ws = (char*)d_ws;
  size_t off = 0;
  auto alloc = [&](size_t bytes) -> char* {
    off = (off + 255) & ~(size_t)255;
    char* p = ws + off;
    off += bytes;
    return p;
  };
  auto cvt = [&](const float* src, int n) -> __bf16* {
    __bf16* dst = (__bf16*)alloc((size_t)n * 2);
    cvt_bf16_kernel<<<(n + WGSIZE - 1) / WGSIZE, WGSIZE, 0, stream>>>(src, dst, n);
    return dst;
  };

  // bf16 weight mirrors
  __bf16* linWb = cvt(linW, 512 * 64);
  __bf16* encW1b[2], *encW2b[2];
  for (int j = 0; j < 2; ++j) {
    encW1b[j] = cvt((const float*)d_in[5 + 6 * j + 2], 1024 * 512);
    encW2b[j] = cvt((const float*)d_in[5 + 6 * j + 4], 512 * 1024);
  }
  __bf16* KWb  = cvt(KW, 512 * 512);
  __bf16* TW1b = cvt(TW1, 1024 * 512);
  __bf16* TW2b = cvt(TW2, 512 * 1024);
  __bf16* predW1b[2], *predW2b[2];
  for (int j = 0; j < 2; ++j) {
    predW1b[j] = cvt((const float*)d_in[26 + 6 * j + 2], 1024 * 512);
    predW2b[j] = cvt((const float*)d_in[26 + 6 * j + 4], 512 * 1024);
  }

  // activations / scratch
  float*  Hc   = (float*)alloc((size_t)MC * 512 * 4);
  float*  kc   = (float*)alloc((size_t)MC * 512 * 4);
  __bf16* kcb  = (__bf16*)alloc((size_t)MC * 512 * 2);
  float*  cksq = (float*)alloc((size_t)MC * 4);
  float*  Hx   = (float*)alloc((size_t)MX * 512 * 4);
  float*  kx   = (float*)alloc((size_t)MX * 512 * 4);
  int*    cidx = (int*)alloc((size_t)MX * CTX * 4);
  float*  Hp   = (float*)alloc((size_t)MP * 512 * 4);

  const size_t smemGemm64  = (size_t)64 * 64 * 2 + 64 * 4;
  const size_t smemGemm512 = (size_t)64 * 512 * 2 + 64 * 4;
  const size_t smemMlp     = (size_t)64 * 512 * 2 + (size_t)64 * 128 * 2;
  const size_t smemTopk    = (size_t)16 * 512 * 2 + (size_t)16 * 1024 * 4 +
                             (size_t)16 * 96 * 4 * 2;
  const size_t smemCtx     = (size_t)(512 + 96 * 3 + 512) * 4 +
                             (size_t)96 * 512 * 2 + (size_t)48 * 128 * 2;

  // encode: 64 -> 512 linear
  gemm_ln_kernel<<<MC / 64, WGSIZE, smemGemm64, stream>>>(
      cand_x, linWb, linb, nullptr, nullptr, Hc, nullptr, nullptr, 64, 0);
  gemm_ln_kernel<<<MX / 64, WGSIZE, smemGemm64, stream>>>(
      x_num, linWb, linb, nullptr, nullptr, Hx, nullptr, nullptr, 64, 0);

  // encoder blocks (block 0: no LN; block 1: LN)
  for (int j = 0; j < 2; ++j) {
    const float* b1 = (const float*)d_in[5 + 6 * j + 3];
    const float* b2 = (const float*)d_in[5 + 6 * j + 5];
    const float* lw = (const float*)d_in[5 + 6 * j + 0];
    const float* lb = (const float*)d_in[5 + 6 * j + 1];
    mlp_block_kernel<<<MC / 64, WGSIZE, smemMlp, stream>>>(
        Hc, encW1b[j], b1, encW2b[j], b2, lw, lb, j == 0 ? 0 : 1, 0);
    mlp_block_kernel<<<MX / 64, WGSIZE, smemMlp, stream>>>(
        Hx, encW1b[j], b1, encW2b[j], b2, lw, lb, j == 0 ? 0 : 1, 0);
  }

  // k-projection (mix LN): candidates also get bf16 mirror + ||k||^2
  gemm_ln_kernel<<<MC / 64, WGSIZE, smemGemm512, stream>>>(
      Hc, KWb, Kb, mixlnw, mixlnb, kc, kcb, cksq, 512, 1);
  gemm_ln_kernel<<<MX / 64, WGSIZE, smemGemm512, stream>>>(
      Hx, KWb, Kb, mixlnw, mixlnb, kx, nullptr, nullptr, 512, 1);

  // top-96 nearest candidates per query
  topk_kernel<<<MX / 16, WGSIZE, smemTopk, stream>>>(kx, kcb, cksq, cidx);

  // context aggregation (softmax + T-MLP + label emb + residual), in-place Hx
  context_kernel<<<MX, WGSIZE, smemCtx, stream>>>(
      kx, kc, cand_y, cidx, TW1b, Tb1, TW2b, labW, labb, Hx);

  // broadcast to K_ENS=8 ensemble rows
  broadcast8_kernel<<<(MP * 512 + WGSIZE - 1) / WGSIZE, WGSIZE, 0, stream>>>(
      Hx, Hp, MP * 512);

  // predictor blocks (per-ensemble LN)
  for (int j = 0; j < 2; ++j) {
    const float* plw = (const float*)d_in[26 + 6 * j + 0];
    const float* plb = (const float*)d_in[26 + 6 * j + 1];
    const float* b1  = (const float*)d_in[26 + 6 * j + 3];
    const float* b2  = (const float*)d_in[26 + 6 * j + 5];
    mlp_block_kernel<<<MP / 64, WGSIZE, smemMlp, stream>>>(
        Hp, predW1b[j], b1, predW2b[j], b2, plw, plb, 1, 1);
  }

  // head
  head_kernel<<<MP / 8, WGSIZE, 0, stream>>>(Hp, hlnw, hlnb, hW, hb,
                                             (float*)d_out);
}
